// LeakyRNN_18880676233540
// MI455X (gfx1250) — matrix operand order repaired and checked by reading, not compiled, not run on hardware
//
#include <hip/hip_runtime.h>
#include <hip/hip_bf16.h>

// Persistent weights-stationary WMMA RNN for gfx1250 (MI455X).
//   grid = 32 blocks x 512 threads (16 wave32 waves).
//   Recurrent GEMM: 512 16x16 tiles -> 1 tile/wave, K=576 -> 18 v_wmma_f32_16x16x32_f16.
//   W_full fragments live in VGPRs for the whole kernel (144 VGPRs/lane).
//   tanh(state) is computed ONCE per element per step (by its owner lane) and
//   stored as f16 in a ping-pong buffer; all A-fragments of the recurrent GEMM
//   and the readout GEMM load it directly -> 32x less v_tanh/v_cvt pressure.
//   One device-wide ticket barrier per timestep.
//
// d_ws layout (needs ~1.6 MB):
//   [0      .. 255]      barrier counter (+pad)
//   [256    .. +512KB]   state buf0 (256x512 f32)
//   [..     .. +512KB]   state buf1 (256x512 f32)
//   [..     .. +256KB]   hT buf0    (256x512 f16 = tanh(state))
//   [..     .. +256KB]   hT buf1

#define T_STEPS 512
#define BATCH   256
#define DSX     32
#define DCX     32
#define HID     512
#define YDIM    64
#define KDIM    576   // DSX + DCX + HID
#define ALPHA   0.5f

#define NBLOCKS  32
#define NTHREADS 512

typedef __attribute__((ext_vector_type(16))) _Float16 v16h;
typedef __attribute__((ext_vector_type(8)))  _Float16 v8h;
typedef __attribute__((ext_vector_type(8)))  float    v8f;

__device__ __forceinline__ float fast_tanh(float x) {
#if __has_builtin(__builtin_amdgcn_tanhf)
    return __builtin_amdgcn_tanhf(x);        // hardware v_tanh_f32 on gfx1250
#else
    return tanhf(x);
#endif
}

// f32 source -> f16 fragment (two 32B-aligned runs of 8 floats).
__device__ __forceinline__ v16h make_frag_f32(const float* __restrict__ p0,
                                              const float* __restrict__ p1) {
    float4 a = ((const float4*)p0)[0];
    float4 b = ((const float4*)p0)[1];
    float4 c = ((const float4*)p1)[0];
    float4 d = ((const float4*)p1)[1];
    float f[16] = {a.x, a.y, a.z, a.w, b.x, b.y, b.z, b.w,
                   c.x, c.y, c.z, c.w, d.x, d.y, d.z, d.w};
    v16h r;
#pragma unroll
    for (int i = 0; i < 16; ++i) r[i] = (_Float16)f[i];
    return r;
}

// f16 source -> fragment (two 16B-aligned runs of 8 halves), pure loads.
__device__ __forceinline__ v16h load_frag_f16(const _Float16* __restrict__ p0,
                                              const _Float16* __restrict__ p1) {
    v8h a = *(const v8h*)p0;
    v8h b = *(const v8h*)p1;
    v16h r;
#pragma unroll
    for (int i = 0; i < 8; ++i) { r[i] = a[i]; r[i + 8] = b[i]; }
    return r;
}

__device__ __forceinline__ v8f wmma16(v16h a, v16h b, v8f c) {
    // (neg_a, A, neg_b, B, c_mod, C, reuse_a, reuse_b)
    return __builtin_amdgcn_wmma_f32_16x16x32_f16(false, a, false, b,
                                                  (short)0, c, false, false);
}

// Monotonic-ticket grid barrier (all NBLOCKS blocks resident by construction).
__device__ __forceinline__ void grid_barrier(unsigned* __restrict__ cnt) {
    __threadfence();
    __syncthreads();
    if (threadIdx.x == 0) {
        unsigned my = __hip_atomic_fetch_add(cnt, 1u, __ATOMIC_ACQ_REL,
                                             __HIP_MEMORY_SCOPE_AGENT) + 1u;
        unsigned target = ((my + NBLOCKS - 1u) / NBLOCKS) * NBLOCKS;
        while (__hip_atomic_load(cnt, __ATOMIC_ACQUIRE,
                                 __HIP_MEMORY_SCOPE_AGENT) < target) {
            __builtin_amdgcn_s_sleep(2);
        }
    }
    __syncthreads();
    __threadfence();
}

// Zero the whole workspace region we use (counter + state + hT buffers).
#define WS_BYTES (256u + 2u*BATCH*HID*4u + 2u*BATCH*HID*2u)

__global__ void init_ws_kernel(unsigned* __restrict__ ws) {
    size_t i = (size_t)blockIdx.x * blockDim.x + threadIdx.x;
    const size_t n = WS_BYTES / 4u;
    for (size_t k = i; k < n; k += (size_t)gridDim.x * blockDim.x) ws[k] = 0u;
}

__global__ void __launch_bounds__(NTHREADS)
leaky_rnn_persistent(const float* __restrict__ s,  const float* __restrict__ c,
                     const float* __restrict__ Wf, const float* __restrict__ b_in,
                     const float* __restrict__ Wo, const float* __restrict__ b_out,
                     float* __restrict__ out, float* __restrict__ stateBufs,
                     _Float16* __restrict__ hTBufs, unsigned* __restrict__ cnt) {
    const int tid  = threadIdx.x;
    const int lane = tid & 31;
    const int half = lane >> 4;     // 0: lanes 0-15, 1: lanes 16-31
    const int l16  = lane & 15;
    const int w    = tid >> 5;      // wave 0..15
    const int bx   = blockIdx.x;    // 0..31
    const int wgM  = bx & 3;        // 4 M-groups of 64 batch rows
    const int wgN  = bx >> 2;       // 8 N-groups of 64 hidden cols

    // ---- Phase-B tile for this wave: (mt, nt) in the 4x4 tile grid of this WG
    const int mt  = w & 3;
    const int nt  = w >> 2;
    const int gmA = wgM * 64 + mt * 16 + l16;   // batch row held by this lane (A layout)
    const int gn  = wgN * 64 + nt * 16 + l16;   // hidden col held by this lane (B/D layout)
    const float binv = b_in[gn];

    // ---- Phase-C (readout) tile: waves 0,1 only -> 64 tiles total
    const int oid = bx * 2 + w;     // valid for w < 2
    const int tm  = oid & 15;       // batch tile
    const int tn  = oid >> 4;       // output tile
    float bov = 0.0f;
    if (w < 2) bov = b_out[tn * 16 + l16];

    float*    buf0 = stateBufs;
    float*    buf1 = stateBufs + (size_t)BATCH * HID;
    _Float16* hT0  = hTBufs;
    _Float16* hT1  = hTBufs + (size_t)BATCH * HID;

    // ---- Preload this wave's W_full B-fragments into registers (stationary).
    // B[k][n] = W_full[n][k] -> two runs of 8 f32 (k = 8h.., 16+8h..) per fragment per lane.
    v16h bfrag[KDIM / 32];
#pragma unroll
    for (int ks = 0; ks < KDIM / 32; ++ks) {
        const float* wr = Wf + (size_t)gn * KDIM + ks * 32 + half * 8;
        bfrag[ks] = make_frag_f32(wr, wr + 16);
    }

#pragma unroll 1
    for (int t = 0; t < T_STEPS; ++t) {
        const float*    bufIn  = (t & 1) ? buf1 : buf0;
        float*          bufOut = (t & 1) ? buf0 : buf1;
        const _Float16* hTIn   = (t & 1) ? hT1  : hT0;
        _Float16*       hTOut  = (t & 1) ? hT0  : hT1;

        // ================= Phase B: recurrent GEMM (all 16 waves) =========
        const float*    srow = s + ((size_t)t * BATCH + gmA) * DSX;
        const float*    crow = c + ((size_t)t * BATCH + gmA) * DCX;
        const _Float16* hrow = hTIn + (size_t)gmA * HID;

        v8f acc = {0.f, 0.f, 0.f, 0.f, 0.f, 0.f, 0.f, 0.f};
#pragma unroll
        for (int ks = 0; ks < KDIM / 32; ++ks) {
            v16h a;
            if (ks == 0) {
                a = make_frag_f32(srow + half * 8, srow + 16 + half * 8);
            } else if (ks == 1) {
                a = make_frag_f32(crow + half * 8, crow + 16 + half * 8);
            } else {
                const _Float16* p = hrow + (ks * 32 - 64);
                a = load_frag_f16(p + half * 8, p + 16 + half * 8);
            }
            acc = wmma16(a, bfrag[ks], acc);
        }

        // Leaky state update; each (m,n) element owned by exactly one lane.
        // C/D layout: VGPR r -> M = 8*half + r, N = l16.
        // Owner also materializes tanh(new_state) as f16 (computed ONCE).
#pragma unroll
        for (int r = 0; r < 8; ++r) {
            const int gm = wgM * 64 + mt * 16 + half * 8 + r;
            const size_t idx = (size_t)gm * HID + gn;
            const float old = bufIn[idx];
            const float nv  = (1.0f - ALPHA) * old + ALPHA * (acc[r] + binv);
            bufOut[idx] = nv;
            hTOut[idx]  = (_Float16)fast_tanh(nv);
        }

        grid_barrier(cnt);   // all of bufOut/hTOut written before anyone reads

        // ================= Phase C: readout GEMM (waves 0,1) ==============
        // out = tanh(new_state) @ Wout^T + b_out ; hTOut IS tanh(new_state).
        if (w < 2) {
            const int am = tm * 16 + l16;
            const _Float16* hr = hTOut + (size_t)am * HID;
            v8f occ = {0.f, 0.f, 0.f, 0.f, 0.f, 0.f, 0.f, 0.f};
#pragma unroll
            for (int ks = 0; ks < HID / 32; ++ks) {
                v16h a = load_frag_f16(hr + ks * 32 + half * 8,
                                       hr + ks * 32 + 16 + half * 8);
                const float* wr = Wo + (size_t)(tn * 16 + l16) * HID
                                     + ks * 32 + half * 8;
                v16h b = make_frag_f32(wr, wr + 16);
                occ = wmma16(a, b, occ);
            }
#pragma unroll
            for (int r = 0; r < 8; ++r) {
                const int m = tm * 16 + half * 8 + r;
                const int n = tn * 16 + l16;
                out[((size_t)t * BATCH + m) * YDIM + n] = occ[r] + bov;
            }
        }
        // No second barrier needed: next step writes the *other* buffers
        // (ping-pong); these buffers are next overwritten only after the
        // following barrier.
    }
}

extern "C" void kernel_launch(void* const* d_in, const int* in_sizes, int n_in,
                              void* d_out, int out_size, void* d_ws, size_t ws_size,
                              hipStream_t stream) {
    const float* s     = (const float*)d_in[0];   // [512,256,32]
    const float* c     = (const float*)d_in[1];   // [512,256,32]
    const float* Wf    = (const float*)d_in[2];   // [512,576]
    const float* b_in  = (const float*)d_in[3];   // [512]
    const float* Wo    = (const float*)d_in[4];   // [64,512]
    const float* b_out = (const float*)d_in[5];   // [64]
    float* out = (float*)d_out;                   // [512,256,64] f32

    unsigned* cnt  = (unsigned*)d_ws;
    float*    state = (float*)((char*)d_ws + 256);
    _Float16* hT    = (_Float16*)((char*)d_ws + 256 + 2ull * BATCH * HID * 4ull);

    init_ws_kernel<<<128, 256, 0, stream>>>((unsigned*)d_ws);
    leaky_rnn_persistent<<<NBLOCKS, NTHREADS, 0, stream>>>(
        s, c, Wf, b_in, Wo, b_out, out, state, hT, cnt);
}
